// ViTRoPENew_40364102648147
// MI455X (gfx1250) — matrix operand order repaired and checked by reading, not compiled, not run on hardware
//
#include <hip/hip_runtime.h>
#include <hip/hip_bf16.h>
#include <math.h>

typedef __attribute__((ext_vector_type(16))) __bf16 v16bf;
typedef __attribute__((ext_vector_type(8)))  __bf16 v8bf;
typedef __attribute__((ext_vector_type(2)))  __bf16 v2bf;
typedef __attribute__((ext_vector_type(8)))  float  v8f;

// ---------------- problem constants ----------------
#define B_     8
#define CH_    3
#define HW_    224
#define P_     16
#define NP_    196      // 14*14 patches
#define M_     197      // tokens (cls + patches)
#define DIM_   512
#define HEADS_ 8
#define DEPTH_ 2
#define FCT_   8
#define JD_    64       // dim/FACTOR
#define TD_    1576     // M_*FACTOR
#define NCLS_  1000
#define PK_    768      // patch pixels * channels

// ---------------- GEMM tiling ----------------
#define TM   64
#define TN   64
#define TKS  32
#define GT   128        // 4 waves of 32
#define LDT  40         // 32 + 8 pad (bf16 elems); rows stay 16B aligned

// C[m,n] = alpha * sum_k A[m,k]*B[k,n] (+bias[n]) (+Res[m,n])
// A fp32 row-major (Mr x Kr). B fp32 row-major (Kr x Nr), or (Nr x Kr) if TRANSB.
// Batched over blockIdx.z: Aoff=(z/aDiv)*sA, Boff=(z%bMod)*sB, Coff=z*sC, Resoff=z*sR.
// Both LDS tiles stored as [row(m|n)][k]: every fragment = two contiguous 16B
// chunks -> ds_load_b128 pairs.
template <bool TRANSB>
__global__ void __launch_bounds__(GT, 2)
gemm_bf16_wmma(const float* __restrict__ Ag, const float* __restrict__ Bg,
               const float* __restrict__ bias, const float* __restrict__ Resg,
               float* __restrict__ Cg,
               int Mr, int Nr, int Kr,
               long long sA, long long sB, long long sC, long long sR,
               int aDiv, int bMod, float alpha)
{
    __shared__ __bf16 As[TM * LDT];   // [m][k]
    __shared__ __bf16 Bs[TN * LDT];   // [n][k]  (transposed at staging)

    const int z = blockIdx.z;
    const float* A = Ag + (long long)(z / aDiv) * sA;
    const float* Bp = Bg + (long long)(z % bMod) * sB;
    float* C = Cg + (long long)z * sC;
    const float* R = Resg ? (Resg + (long long)z * sR) : nullptr;

    const int m0 = blockIdx.y * TM;
    const int n0 = blockIdx.x * TN;
    const int t = threadIdx.x;
    const int lane = t & 31;
    const int wave = t >> 5;
    const int wm = (wave >> 1) * 32;   // wave's 32x32 subtile
    const int wn = (wave & 1) * 32;
    const int lrow = lane & 15;
    const int lh   = lane >> 4;

    v8f acc[2][2];
    const v8f vz = {0.f,0.f,0.f,0.f,0.f,0.f,0.f,0.f};
    acc[0][0] = vz; acc[0][1] = vz; acc[1][0] = vz; acc[1][1] = vz;

    // ---- staging helpers (branch-free inner bodies) ----
    auto stageA_fast = [&](int k0) {
#pragma unroll
        for (int i = 0; i < 8; ++i) {
            int p = t + i * GT;
            int row = p >> 4, cp = (p & 15) * 2;
            const float* src = A + (long long)(m0 + row) * Kr + k0 + cp;
            float v0 = src[0], v1 = src[1];
            v2bf pv; pv[0] = (__bf16)v0; pv[1] = (__bf16)v1;
            *(v2bf*)(As + row * LDT + cp) = pv;
        }
    };
    auto stageA_edge = [&](int k0) {      // clamp + multiply-mask, loads unconditional
#pragma unroll
        for (int i = 0; i < 8; ++i) {
            int p = t + i * GT;
            int row = p >> 4, cp = (p & 15) * 2;
            int gm = m0 + row, gk0i = k0 + cp, gk1i = gk0i + 1;
            int gmc = gm < Mr ? gm : (Mr - 1);
            int k0c = gk0i < Kr ? gk0i : (Kr - 1);
            int k1c = gk1i < Kr ? gk1i : (Kr - 1);
            const float* rp = A + (long long)gmc * Kr;
            float v0 = rp[k0c] * (float)((gm < Mr) & (gk0i < Kr));
            float v1 = rp[k1c] * (float)((gm < Mr) & (gk1i < Kr));
            v2bf pv; pv[0] = (__bf16)v0; pv[1] = (__bf16)v1;
            *(v2bf*)(As + row * LDT + cp) = pv;
        }
    };
    auto stageB_fast = [&](int k0) {
        if (TRANSB) {
            // k fastest: lanes read contiguous k within an n row
#pragma unroll
            for (int i = 0; i < 8; ++i) {
                int p = t + i * GT;
                int n = p >> 4, cp = (p & 15) * 2;
                const float* src = Bp + (long long)(n0 + n) * Kr + k0 + cp;
                float v0 = src[0], v1 = src[1];
                v2bf pv; pv[0] = (__bf16)v0; pv[1] = (__bf16)v1;
                *(v2bf*)(Bs + n * LDT + cp) = pv;
            }
        } else {
            // n fastest: consecutive lanes -> consecutive n (coalesced)
#pragma unroll
            for (int i = 0; i < 8; ++i) {
                int p = t + i * GT;
                int n = p & 63, kp = p >> 6, kkk = kp * 2;
                long long base = (long long)(k0 + kkk) * Nr + n0 + n;
                float v0 = Bp[base], v1 = Bp[base + Nr];
                v2bf pv; pv[0] = (__bf16)v0; pv[1] = (__bf16)v1;
                *(v2bf*)(Bs + n * LDT + kkk) = pv;
            }
        }
    };
    auto stageB_edge = [&](int k0) {
        if (TRANSB) {
#pragma unroll
            for (int i = 0; i < 8; ++i) {
                int p = t + i * GT;
                int n = p >> 4, cp = (p & 15) * 2;
                int gn = n0 + n, gk0i = k0 + cp, gk1i = gk0i + 1;
                int gnc = gn < Nr ? gn : (Nr - 1);
                int k0c = gk0i < Kr ? gk0i : (Kr - 1);
                int k1c = gk1i < Kr ? gk1i : (Kr - 1);
                const float* rp = Bp + (long long)gnc * Kr;
                float v0 = rp[k0c] * (float)((gn < Nr) & (gk0i < Kr));
                float v1 = rp[k1c] * (float)((gn < Nr) & (gk1i < Kr));
                v2bf pv; pv[0] = (__bf16)v0; pv[1] = (__bf16)v1;
                *(v2bf*)(Bs + n * LDT + cp) = pv;
            }
        } else {
#pragma unroll
            for (int i = 0; i < 8; ++i) {
                int p = t + i * GT;
                int n = p & 63, kp = p >> 6, kkk = kp * 2;
                int gn = n0 + n, gk0i = k0 + kkk, gk1i = gk0i + 1;
                int gnc = gn < Nr ? gn : (Nr - 1);
                int k0c = gk0i < Kr ? gk0i : (Kr - 1);
                int k1c = gk1i < Kr ? gk1i : (Kr - 1);
                float v0 = Bp[(long long)k0c * Nr + gnc] * (float)((gn < Nr) & (gk0i < Kr));
                float v1 = Bp[(long long)k1c * Nr + gnc] * (float)((gn < Nr) & (gk1i < Kr));
                v2bf pv; pv[0] = (__bf16)v0; pv[1] = (__bf16)v1;
                *(v2bf*)(Bs + n * LDT + kkk) = pv;
            }
        }
    };
    auto mma = [&]() {
        v16bf afr[2], bfr[2];
#pragma unroll
        for (int r = 0; r < 2; ++r) {
            const __bf16* base = As + (wm + r * 16 + lrow) * LDT;
            v8bf lo = *(const v8bf*)(base + 8 * lh);
            v8bf hi = *(const v8bf*)(base + 16 + 8 * lh);
            afr[r] = __builtin_shufflevector(lo, hi, 0,1,2,3,4,5,6,7,8,9,10,11,12,13,14,15);
        }
#pragma unroll
        for (int c = 0; c < 2; ++c) {
            const __bf16* base = Bs + (wn + c * 16 + lrow) * LDT;
            v8bf lo = *(const v8bf*)(base + 8 * lh);
            v8bf hi = *(const v8bf*)(base + 16 + 8 * lh);
            bfr[c] = __builtin_shufflevector(lo, hi, 0,1,2,3,4,5,6,7,8,9,10,11,12,13,14,15);
        }
#pragma unroll
        for (int r = 0; r < 2; ++r)
#pragma unroll
            for (int c = 0; c < 2; ++c)
                acc[r][c] = __builtin_amdgcn_wmma_f32_16x16x32_bf16(
                    false, afr[r], false, bfr[c], (short)0, acc[r][c], false, false);
    };

    const int kFull = (Kr / TKS) * TKS;
    const bool fullMN = (m0 + TM <= Mr) && (n0 + TN <= Nr);

    if (fullMN) {
        for (int k0 = 0; k0 < kFull; k0 += TKS) {       // branch-free steady state
            stageA_fast(k0); stageB_fast(k0);
            __syncthreads(); mma(); __syncthreads();
        }
        if (kFull < Kr) {                                // K tail (once)
            stageA_edge(kFull); stageB_edge(kFull);
            __syncthreads(); mma(); __syncthreads();
        }
    } else {
        for (int k0 = 0; k0 < Kr; k0 += TKS) {
            stageA_edge(k0); stageB_edge(k0);
            __syncthreads(); mma(); __syncthreads();
        }
    }

    // ---- store C (16x16 f32 layout: VGPR i -> M = i + 8*laneHalf, N = lane&15) ----
#pragma unroll
    for (int r = 0; r < 2; ++r) {
#pragma unroll
        for (int c = 0; c < 2; ++c) {
            int gn = n0 + wn + c * 16 + lrow;
#pragma unroll
            for (int i = 0; i < 8; ++i) {
                int gm = m0 + wm + r * 16 + i + lh * 8;
                if (gm < Mr && gn < Nr) {
                    float vv = acc[r][c][i] * alpha;
                    if (bias) vv += bias[gn];
                    if (R)    vv += R[(long long)gm * Nr + gn];
                    C[(long long)gm * Nr + gn] = vv;
                }
            }
        }
    }
}

// ---------------- elementwise kernels ----------------

__global__ void im2col_k(const float* __restrict__ img, float* __restrict__ out) {
    long long idx = (long long)blockIdx.x * blockDim.x + threadIdx.x;
    long long total = (long long)B_ * NP_ * PK_;
    if (idx >= total) return;
    int f = (int)(idx % PK_);
    int n = (int)((idx / PK_) % NP_);
    int b = (int)(idx / ((long long)PK_ * NP_));
    int c = f % CH_;
    int pxy = f / CH_;
    int px = pxy % P_, py = pxy / P_;
    int wx = n % 14, hy = n / 14;
    out[idx] = img[(((long long)b * CH_ + c) * HW_ + hy * P_ + py) * HW_ + wx * P_ + px];
}

__global__ void assemble_x_k(const float* __restrict__ pout, const float* __restrict__ cls,
                             float* __restrict__ x) {
    long long idx = (long long)blockIdx.x * blockDim.x + threadIdx.x;
    long long total = (long long)B_ * M_ * DIM_;
    if (idx >= total) return;
    int d = (int)(idx % DIM_);
    int m = (int)((idx / DIM_) % M_);
    int b = (int)(idx / ((long long)DIM_ * M_));
    x[idx] = (m == 0) ? cls[d] : pout[((long long)b * NP_ + (m - 1)) * DIM_ + d];
}

// in-place RMS norm over (m,d) per batch: x = scale * x * rsqrt(mean(x^2))
__global__ void __launch_bounds__(256)
rmsnorm_k(float* __restrict__ x, const float* __restrict__ scale, int mrows, int d) {
    __shared__ float red[256];
    __shared__ float invS;
    int b = blockIdx.x, t = threadIdx.x;
    long long n = (long long)mrows * d;
    float* xb = x + (long long)b * n;
    float s = 0.0f;
    for (long long i = t; i < n; i += 256) { float v = xb[i]; s += v * v; }
    red[t] = s; __syncthreads();
    for (int o = 128; o > 0; o >>= 1) { if (t < o) red[t] += red[t + o]; __syncthreads(); }
    if (t == 0) invS = rsqrtf(red[0] / (float)n);
    __syncthreads();
    float inv = invS;
    for (long long i = t; i < n; i += 256) xb[i] = scale[i] * xb[i] * inv;
}

// interleaved RoPE in place on (nmat, npos, d); tables computed on the fly
__global__ void rope_k(float* __restrict__ x, int nmat, int npos, int d) {
    int half = d >> 1;
    long long idx = (long long)blockIdx.x * blockDim.x + threadIdx.x;
    long long total = (long long)nmat * npos * half;
    if (idx >= total) return;
    int i = (int)(idx % half);
    int pos = (int)((idx / half) % npos);
    long long mat = idx / ((long long)half * npos);
    float ex = -2.0f * ((float)i - 1.0f) / (float)d;   // reference uses (i-1)
    float theta = __powf(10000.0f, ex);
    float ang = (float)pos * theta;
    float cs = __cosf(ang), sn = __sinf(ang);
    float* p = x + (mat * (long long)npos + pos) * d + 2 * i;
    float xe = p[0], xo = p[1];
    p[0] =  xe * cs + xo * sn;
    p[1] = -xe * sn + xo * cs;
}

__global__ void __launch_bounds__(128)
softmax_k(float* __restrict__ s, int L) {
    __shared__ float red[128];
    __shared__ float mS, sS;
    float* row = s + (long long)blockIdx.x * L;
    int t = threadIdx.x;
    float mx = -1e30f;
    for (int i = t; i < L; i += 128) mx = fmaxf(mx, row[i]);
    red[t] = mx; __syncthreads();
    for (int o = 64; o > 0; o >>= 1) { if (t < o) red[t] = fmaxf(red[t], red[t + o]); __syncthreads(); }
    if (t == 0) mS = red[0]; __syncthreads();
    float m = mS, sum = 0.0f;
    for (int i = t; i < L; i += 128) { float e = __expf(row[i] - m); row[i] = e; sum += e; }
    red[t] = sum; __syncthreads();
    for (int o = 64; o > 0; o >>= 1) { if (t < o) red[t] += red[t + o]; __syncthreads(); }
    if (t == 0) sS = red[0]; __syncthreads();
    float inv = 1.0f / sS;
    for (int i = t; i < L; i += 128) row[i] *= inv;
}

// (B,H,M,D) -> (B,M,H*D)
__global__ void o_transpose_k(const float* __restrict__ O, float* __restrict__ o2) {
    long long idx = (long long)blockIdx.x * blockDim.x + threadIdx.x;
    long long total = (long long)B_ * M_ * HEADS_ * DIM_;
    if (idx >= total) return;
    int e = (int)(idx % DIM_);
    int h = (int)((idx / DIM_) % HEADS_);
    int mm = (int)((idx / ((long long)DIM_ * HEADS_)) % M_);
    int b = (int)(idx / ((long long)DIM_ * HEADS_ * M_));
    o2[idx] = O[(((long long)b * HEADS_ + h) * M_ + mm) * DIM_ + e];
}

// y[b][j][f*M+mm] = x[b][mm][j*F+f]
__global__ void ypack_k(const float* __restrict__ x, float* __restrict__ y) {
    long long idx = (long long)blockIdx.x * blockDim.x + threadIdx.x;
    long long total = (long long)B_ * JD_ * TD_;
    if (idx >= total) return;
    int t1 = (int)(idx % TD_);
    int j = (int)((idx / TD_) % JD_);
    int b = (int)(idx / ((long long)TD_ * JD_));
    int f = t1 / M_, mm = t1 % M_;
    y[idx] = x[((long long)b * M_ + mm) * DIM_ + j * FCT_ + f];
}

__global__ void yunpack_k(const float* __restrict__ y, float* __restrict__ x) {
    long long idx = (long long)blockIdx.x * blockDim.x + threadIdx.x;
    long long total = (long long)B_ * M_ * DIM_;
    if (idx >= total) return;
    int d = (int)(idx % DIM_);
    int mm = (int)((idx / DIM_) % M_);
    int b = (int)(idx / ((long long)DIM_ * M_));
    int j = d / FCT_, f = d % FCT_;
    x[idx] = y[((long long)b * JD_ + j) * TD_ + f * M_ + mm];
}

// LayerNorm of cls row -> out (B, DIM)
__global__ void __launch_bounds__(256)
ln_cls_k(const float* __restrict__ x, const float* __restrict__ g,
         const float* __restrict__ bb, float* __restrict__ out) {
    __shared__ float red[256];
    __shared__ float muS, ivS;
    int b = blockIdx.x, t = threadIdx.x;
    const float* r = x + (long long)b * M_ * DIM_;   // row m=0
    float s = 0.0f;
    for (int i = t; i < DIM_; i += 256) s += r[i];
    red[t] = s; __syncthreads();
    for (int o = 128; o > 0; o >>= 1) { if (t < o) red[t] += red[t + o]; __syncthreads(); }
    if (t == 0) muS = red[0] / (float)DIM_;
    __syncthreads();
    float mu = muS;
    s = 0.0f;
    for (int i = t; i < DIM_; i += 256) { float d = r[i] - mu; s += d * d; }
    red[t] = s; __syncthreads();
    for (int o = 128; o > 0; o >>= 1) { if (t < o) red[t] += red[t + o]; __syncthreads(); }
    if (t == 0) ivS = rsqrtf(red[0] / (float)DIM_ + 1e-5f);
    __syncthreads();
    float iv = ivS;
    for (int i = t; i < DIM_; i += 256)
        out[(long long)b * DIM_ + i] = (r[i] - mu) * iv * g[i] + bb[i];
}

// ---------------- host orchestration ----------------
extern "C" void kernel_launch(void* const* d_in, const int* in_sizes, int n_in,
                              void* d_out, int out_size, void* d_ws, size_t ws_size,
                              hipStream_t stream) {
    const float* img      = (const float*)d_in[0];
    const float* patch_W  = (const float*)d_in[1];
    const float* patch_b  = (const float*)d_in[2];
    const float* cls_tok  = (const float*)d_in[3];
    const float* rms_sc   = (const float*)d_in[4];
    const float* Wq       = (const float*)d_in[5];
    const float* Wk       = (const float*)d_in[6];
    const float* Wv       = (const float*)d_in[7];
    const float* attn_W   = (const float*)d_in[8];
    const float* attn_b   = (const float*)d_in[9];
    const float* tWq      = (const float*)d_in[10];
    const float* tWk      = (const float*)d_in[11];
    const float* tWv      = (const float*)d_in[12];
    const float* ln_g     = (const float*)d_in[13];
    const float* ln_b     = (const float*)d_in[14];
    const float* head_W   = (const float*)d_in[15];
    const float* head_b   = (const float*)d_in[16];
    float* outp = (float*)d_out;

    const long long XSZ   = (long long)B_ * M_ * DIM_;            // 806,912
    const long long QSZ   = (long long)B_ * HEADS_ * M_ * DIM_;   // 6,455,296
    const long long SSZ   = (long long)B_ * HEADS_ * M_ * M_;     // 2,483,776
    const long long MD    = (long long)M_ * DIM_;
    const long long JDTD  = (long long)JD_ * TD_;

    float* wsf = (float*)d_ws;
    float* x   = wsf;
    float* q   = x + XSZ;
    float* kk  = q + QSZ;
    float* v   = kk + QSZ;
    float* S   = v + QSZ;
    float* O   = S + SSZ;
    float* o2  = O + QSZ;
    float* clsn = o2 + QSZ;
    // aliases (lifetime-disjoint)
    float* ximg = q;     // B*NP*768 (4.8MB < 25.8MB)
    float* pout = kk;    // B*NP*512
    float* y  = q;       // token stage
    float* tq = kk;
    float* tk = v;
    float* tv = O;
    float* S2 = S;

    auto gemm = [&](bool transB, const float* A, const float* Bm, const float* bias,
                    const float* Res, float* C, int Mr, int Nr, int Kr, int nz,
                    long long sA, long long sB, long long sC, long long sR,
                    int aDiv, int bMod, float alpha) {
        dim3 grid((Nr + TN - 1) / TN, (Mr + TM - 1) / TM, nz);
        if (transB)
            gemm_bf16_wmma<true><<<grid, GT, 0, stream>>>(A, Bm, bias, Res, C, Mr, Nr, Kr,
                                                          sA, sB, sC, sR, aDiv, bMod, alpha);
        else
            gemm_bf16_wmma<false><<<grid, GT, 0, stream>>>(A, Bm, bias, Res, C, Mr, Nr, Kr,
                                                           sA, sB, sC, sR, aDiv, bMod, alpha);
    };
    auto nb = [](long long n) { return (int)((n + 255) / 256); };

    const int BIG = 1 << 30;
    const float sc_sp = 1.0f / sqrtf((float)DIM_);
    const float sc_tk = 1.0f / sqrtf((float)TD_);

    // ---- patch embed ----
    im2col_k<<<nb((long long)B_ * NP_ * PK_), 256, 0, stream>>>(img, ximg);
    gemm(false, ximg, patch_W, patch_b, nullptr, pout, B_ * NP_, DIM_, PK_, 1,
         0, 0, 0, 0, 1, 1, 1.0f);
    assemble_x_k<<<nb(XSZ), 256, 0, stream>>>(pout, cls_tok, x);

    for (int l = 0; l < DEPTH_; ++l) {
        const float* rs  = rms_sc + (long long)l * M_ * DIM_;
        const float* wql = Wq + (long long)l * HEADS_ * DIM_ * DIM_;
        const float* wkl = Wk + (long long)l * HEADS_ * DIM_ * DIM_;
        const float* wvl = Wv + (long long)l * HEADS_ * DIM_ * DIM_;
        const float* awl = attn_W + (long long)l * HEADS_ * DIM_ * DIM_;
        const float* abl = attn_b + (long long)l * DIM_;
        const float* twq = tWq + (long long)l * TD_ * TD_;
        const float* twk = tWk + (long long)l * TD_ * TD_;
        const float* twv = tWv + (long long)l * TD_ * TD_;

        // x = rmsnorm(x)
        rmsnorm_k<<<B_, 256, 0, stream>>>(x, rs, M_, DIM_);

        // q/k/v[b,h] = x[b] @ W[h]   (z = b*H + h)
        gemm(false, x, wql, nullptr, nullptr, q, M_, DIM_, DIM_, B_ * HEADS_,
             MD, (long long)DIM_ * DIM_, MD, 0, HEADS_, HEADS_, 1.0f);
        gemm(false, x, wkl, nullptr, nullptr, kk, M_, DIM_, DIM_, B_ * HEADS_,
             MD, (long long)DIM_ * DIM_, MD, 0, HEADS_, HEADS_, 1.0f);
        gemm(false, x, wvl, nullptr, nullptr, v, M_, DIM_, DIM_, B_ * HEADS_,
             MD, (long long)DIM_ * DIM_, MD, 0, HEADS_, HEADS_, 1.0f);

        rope_k<<<nb((long long)B_ * HEADS_ * M_ * (DIM_ / 2)), 256, 0, stream>>>(q, B_ * HEADS_, M_, DIM_);
        rope_k<<<nb((long long)B_ * HEADS_ * M_ * (DIM_ / 2)), 256, 0, stream>>>(kk, B_ * HEADS_, M_, DIM_);

        // S = (q @ k^T) * 1/sqrt(d)
        gemm(true, q, kk, nullptr, nullptr, S, M_, M_, DIM_, B_ * HEADS_,
             MD, MD, (long long)M_ * M_, 0, 1, BIG, sc_sp);
        softmax_k<<<B_ * HEADS_ * M_, 128, 0, stream>>>(S, M_);
        // O = P @ v
        gemm(false, S, v, nullptr, nullptr, O, M_, DIM_, M_, B_ * HEADS_,
             (long long)M_ * M_, MD, MD, 0, 1, BIG, 1.0f);

        o_transpose_k<<<nb(QSZ), 256, 0, stream>>>(O, o2);
        // x = x + o2 @ attn_W + attn_b   (in-place residual)
        gemm(false, o2, awl, abl, x, x, M_, DIM_, HEADS_ * DIM_, B_,
             (long long)M_ * HEADS_ * DIM_, 0, MD, MD, 1, 1, 1.0f);

        // x = rmsnorm(x)
        rmsnorm_k<<<B_, 256, 0, stream>>>(x, rs, M_, DIM_);

        // token-axis attention
        ypack_k<<<nb((long long)B_ * JDTD), 256, 0, stream>>>(x, y);
        gemm(false, y, twq, nullptr, nullptr, tq, JD_, TD_, TD_, B_,
             JDTD, 0, JDTD, 0, 1, 1, 1.0f);
        gemm(false, y, twk, nullptr, nullptr, tk, JD_, TD_, TD_, B_,
             JDTD, 0, JDTD, 0, 1, 1, 1.0f);
        gemm(false, y, twv, nullptr, nullptr, tv, JD_, TD_, TD_, B_,
             JDTD, 0, JDTD, 0, 1, 1, 1.0f);
        rope_k<<<nb((long long)B_ * JD_ * (TD_ / 2)), 256, 0, stream>>>(tq, B_, JD_, TD_);
        rope_k<<<nb((long long)B_ * JD_ * (TD_ / 2)), 256, 0, stream>>>(tk, B_, JD_, TD_);
        gemm(true, tq, tk, nullptr, nullptr, S2, JD_, JD_, TD_, B_,
             JDTD, JDTD, (long long)JD_ * JD_, 0, 1, BIG, sc_tk);
        softmax_k<<<B_ * JD_, 128, 0, stream>>>(S2, JD_);
        // y = y + P2 @ tv   (in-place residual)
        gemm(false, S2, tv, nullptr, y, y, JD_, TD_, JD_, B_,
             (long long)JD_ * JD_, JDTD, JDTD, JDTD, 1, BIG, 1.0f);
        yunpack_k<<<nb(XSZ), 256, 0, stream>>>(y, x);
    }

    // ---- head ----
    ln_cls_k<<<B_, 256, 0, stream>>>(x, ln_g, ln_b, clsn);
    gemm(false, clsn, head_W, head_b, nullptr, outp, B_, NCLS_, DIM_, 1,
         0, 0, 0, 0, 1, 1, 1.0f);
}
